// WanSelfAttention_36481452212231
// MI455X (gfx1250) — matrix operand order repaired and checked by reading, not compiled, not run on hardware
//
#include <hip/hip_runtime.h>
#include <hip/hip_bf16.h>
#include <math.h>

typedef __attribute__((ext_vector_type(16))) _Float16 v16h;
typedef __attribute__((ext_vector_type(8)))  _Float16 v8h;
typedef __attribute__((ext_vector_type(4)))  _Float16 v4h;
typedef __attribute__((ext_vector_type(8)))  float    v8f;

#define NHEADS 16

// Load a 16-half WMMA fragment from two 16-byte runs.
__device__ __forceinline__ v16h load_frag(const _Float16* p0, const _Float16* p1) {
    union { v16h v; v8h h[2]; } u;
    u.h[0] = *reinterpret_cast<const v8h*>(p0);
    u.h[1] = *reinterpret_cast<const v8h*>(p1);
    return u.v;
}

// ---------------- f32 -> f16 conversion (vectorized) ----------------
__global__ void conv_f16(const float* __restrict__ s, _Float16* __restrict__ d, int n) {
    int i = (blockIdx.x * 256 + threadIdx.x) * 4;
    if (i < n) {
        float4 f = *reinterpret_cast<const float4*>(s + i);
        v4h h; h.x = (_Float16)f.x; h.y = (_Float16)f.y;
               h.z = (_Float16)f.z; h.w = (_Float16)f.w;
        *reinterpret_cast<v4h*>(d + i) = h;
    }
}

// Load one k-group of fragments: 2 A tiles (rows m, m+16) and 4 B tiles.
__device__ __forceinline__ void load_group(
        const _Float16* x0, const _Float16* x1,
        const _Float16* w0, const _Float16* w1,
        const _Float16* w2, const _Float16* w3,
        int kb, int half, v16h a[2], v16h b[4]) {
    a[0] = load_frag(x0 + kb + half * 8, x0 + kb + 16 + half * 8);
    a[1] = load_frag(x1 + kb + half * 8, x1 + kb + 16 + half * 8);
    b[0] = load_frag(w0 + kb + half * 8, w0 + kb + 16 + half * 8);
    b[1] = load_frag(w1 + kb + half * 8, w1 + kb + 16 + half * 8);
    b[2] = load_frag(w2 + kb + half * 8, w2 + kb + 16 + half * 8);
    b[3] = load_frag(w3 + kb + half * 8, w3 + kb + 16 + half * 8);
}

// ---------------- WMMA GEMM: Y = X @ W^T + bias ----------------
// X: MxK f16 row-major.  W: NxK f16 row-major.  Y: MxN f32.
// block = 128 (4 waves); wave computes 32x64; block tile 32x256.
// Double-buffered k-loop (ping-pong fragment groups, compile-time indices).
__global__ __launch_bounds__(128) void gemm_wmma(
        const _Float16* __restrict__ X, const _Float16* __restrict__ W,
        const float* __restrict__ bias, float* __restrict__ Y,
        int M, int N, int K) {
    const int wave = threadIdx.x >> 5;
    const int lane = threadIdx.x & 31;
    const int n0   = lane & 15;
    const int half = lane >> 4;
    const int mb   = blockIdx.x * 32;
    const int nb   = blockIdx.y * 256 + wave * 64;

    v8f acc[2][4] = {{v8f{}, v8f{}, v8f{}, v8f{}}, {v8f{}, v8f{}, v8f{}, v8f{}}};

    const _Float16* x0 = X + (size_t)(mb + n0) * K;
    const _Float16* x1 = X + (size_t)(mb + 16 + n0) * K;
    const _Float16* w0 = W + (size_t)(nb + n0) * K;
    const _Float16* w1 = W + (size_t)(nb + 16 + n0) * K;
    const _Float16* w2 = W + (size_t)(nb + 32 + n0) * K;
    const _Float16* w3 = W + (size_t)(nb + 48 + n0) * K;

    v16h aA[2], bA[4], aB[2], bB[4];
    load_group(x0, x1, w0, w1, w2, w3, 0, half, aA, bA);

    for (int kb = 0; kb < K; kb += 64) {
        // prefetch second half of this 64-wide k-chunk while group A computes
        load_group(x0, x1, w0, w1, w2, w3, kb + 32, half, aB, bB);
#pragma unroll
        for (int t = 0; t < 4; ++t)
            acc[0][t] = __builtin_amdgcn_wmma_f32_16x16x32_f16(
                false, aA[0], false, bA[t], (short)0, acc[0][t], false, false);
#pragma unroll
        for (int t = 0; t < 4; ++t)
            acc[1][t] = __builtin_amdgcn_wmma_f32_16x16x32_f16(
                false, aA[1], false, bA[t], (short)0, acc[1][t], false, false);

        if (kb + 64 < K)
            load_group(x0, x1, w0, w1, w2, w3, kb + 64, half, aA, bA);
#pragma unroll
        for (int t = 0; t < 4; ++t)
            acc[0][t] = __builtin_amdgcn_wmma_f32_16x16x32_f16(
                false, aB[0], false, bB[t], (short)0, acc[0][t], false, false);
#pragma unroll
        for (int t = 0; t < 4; ++t)
            acc[1][t] = __builtin_amdgcn_wmma_f32_16x16x32_f16(
                false, aB[1], false, bB[t], (short)0, acc[1][t], false, false);
    }

#pragma unroll
    for (int i = 0; i < 2; ++i) {
        const int rowBase = mb + i * 16 + half * 8;
#pragma unroll
        for (int t = 0; t < 4; ++t) {
            float bv = bias ? bias[nb + t * 16 + n0] : 0.0f;
#pragma unroll
            for (int r = 0; r < 8; ++r)
                Y[(size_t)(rowBase + r) * N + nb + t * 16 + n0] = acc[i][t][r] + bv;
        }
    }
}

// ---------------- RMSNorm + RoPE epilogue (Q and K) ----------------
__global__ __launch_bounds__(256) void rms_rope(
        const float* __restrict__ Y, const float* __restrict__ g,
        const float* __restrict__ fc, const float* __restrict__ fs,
        _Float16* __restrict__ dst, int S, int Dm, int HD,
        const int* __restrict__ fptr, const int* __restrict__ hptr,
        const int* __restrict__ wptr) {
    const int s = blockIdx.x;
    const int per = Dm / 256;            // 8 for Dm=2048
    __shared__ float red[8];
    const float* row = Y + (size_t)s * Dm;
    const int base = threadIdx.x * per;

    float v[16];
    float ss = 0.0f;
    for (int i = 0; i < per; ++i) { v[i] = row[base + i]; ss += v[i] * v[i]; }
#pragma unroll
    for (int m = 16; m >= 1; m >>= 1) ss += __shfl_xor(ss, m, 32);
    if ((threadIdx.x & 31) == 0) red[threadIdx.x >> 5] = ss;
    __syncthreads();
    float tot = 0.0f;
    for (int i = 0; i < 8; ++i) tot += red[i];
    const float rms = rsqrtf(tot / (float)Dm + 1e-6f);

    const int c  = HD >> 1;
    const int c2 = c / 3;
    const int c1 = c - 2 * c2;
    const int hdim = *hptr, wdim = *wptr;
    (void)fptr;
    const int pf = s / (hdim * wdim);
    const int ph = (s / wdim) % hdim;
    const int pw = s % wdim;

    for (int i = 0; i < per; i += 2) {
        const int col = base + i;
        const float a  = v[i]     * rms * g[col];
        const float bq = v[i + 1] * rms * g[col + 1];
        const int head = col / HD;
        const int dloc = col % HD;
        const int j    = dloc >> 1;
        const int pidx = (j < c1) ? pf : ((j < c1 + c2) ? ph : pw);
        const float cs = fc[pidx * c + j];
        const float sn = fs[pidx * c + j];
        const float o0 = a * cs - bq * sn;
        const float o1 = a * sn + bq * cs;
        const size_t off = ((size_t)head * S + s) * HD + dloc;
        dst[off]     = (_Float16)o0;
        dst[off + 1] = (_Float16)o1;
    }
}

// ---------------- V transpose: SxD f32 -> [D][S] f16 ----------------
__global__ __launch_bounds__(256) void transpose_v(
        const float* __restrict__ src, _Float16* __restrict__ dst, int S, int Dm) {
    __shared__ _Float16 tile[32][33];
    const int sx = blockIdx.x * 32;
    const int dx = blockIdx.y * 32;
    const int tx = threadIdx.x & 31;
    const int ty = threadIdx.x >> 5;
    for (int yy = ty; yy < 32; yy += 8)
        tile[yy][tx] = (_Float16)src[(size_t)(sx + yy) * Dm + dx + tx];
    __syncthreads();
    for (int yy = ty; yy < 32; yy += 8)
        dst[(size_t)(dx + yy) * S + sx + tx] = tile[tx][yy];
}

// ---------------- Flash attention (one wave per head x 16-query block) ----------------
// qh, kh: [H][S][128] f16.  vt: [H][128][S] f16.  aoh: [S][Dm] f16.
__global__ __launch_bounds__(128) void attn_wmma(
        const _Float16* __restrict__ qh, const _Float16* __restrict__ kh,
        const _Float16* __restrict__ vt, _Float16* __restrict__ aoh,
        int S, int Dm, const int* __restrict__ seqlens, float scale) {
    __shared__ _Float16 plds[4][16 * 32];
    const int wave = threadIdx.x >> 5;
    const int lane = threadIdx.x & 31;
    const int n0   = lane & 15;
    const int half = lane >> 4;
    const int head = blockIdx.x;
    const int qb   = (blockIdx.y * 4 + wave) * 16;
    const int HD   = 128;
    const int seqlen = seqlens[0];

    const _Float16* Qh = qh + (size_t)head * S * HD;
    const _Float16* Kh = kh + (size_t)head * S * HD;
    const _Float16* Vt = vt + (size_t)head * HD * S;
    _Float16* P = plds[wave];

    // Preload Q fragments (16 rows x 128 = 4 K-chunks of 32)
    v16h aq[4];
    const _Float16* qrow = Qh + (size_t)(qb + n0) * HD;
#pragma unroll
    for (int c = 0; c < 4; ++c)
        aq[c] = load_frag(qrow + c * 32 + half * 8, qrow + c * 32 + 16 + half * 8);

    v8f o[8] = {v8f{}, v8f{}, v8f{}, v8f{}, v8f{}, v8f{}, v8f{}, v8f{}};
    float mi[8], li[8];
#pragma unroll
    for (int r = 0; r < 8; ++r) { mi[r] = -3.0e38f; li[r] = 0.0f; }

    for (int kb = 0; kb < S; kb += 32) {
        // ---- issue ALL K-fragment loads, then ALL V-fragment loads.
        // S-WMMAs only need the K group, so the V loads stay in flight and
        // overlap the WMMAs + softmax VALU work below.
        v16h bk[8];
#pragma unroll
        for (int t = 0; t < 2; ++t)
#pragma unroll
            for (int c = 0; c < 4; ++c) {
                const _Float16* krow =
                    Kh + (size_t)(kb + t * 16 + n0) * HD + c * 32 + half * 8;
                bk[t * 4 + c] = load_frag(krow, krow + 16);
            }
        v16h bv[8];
#pragma unroll
        for (int t2 = 0; t2 < 8; ++t2) {
            const _Float16* vrow = Vt + (size_t)(t2 * 16 + n0) * S + kb + half * 8;
            bv[t2] = load_frag(vrow, vrow + 16);
        }

        // ---- S tiles: scores for keys [kb, kb+32) ----
        v8f st[2] = {v8f{}, v8f{}};
#pragma unroll
        for (int t = 0; t < 2; ++t) {
#pragma unroll
            for (int c = 0; c < 4; ++c)
                st[t] = __builtin_amdgcn_wmma_f32_16x16x32_f16(
                    false, aq[c], false, bk[t * 4 + c], (short)0, st[t], false, false);
            const bool ok = (kb + t * 16 + n0) < seqlen;
#pragma unroll
            for (int r = 0; r < 8; ++r)
                st[t][r] = ok ? st[t][r] * scale : -3.0e38f;
        }

        // ---- online softmax per row ----
        float pr0[8], pr1[8];
#pragma unroll
        for (int r = 0; r < 8; ++r) {
            float vmax = fmaxf(st[0][r], st[1][r]);
#pragma unroll
            for (int m = 8; m >= 1; m >>= 1) vmax = fmaxf(vmax, __shfl_xor(vmax, m, 32));
            const float newm = fmaxf(mi[r], vmax);
            const float corr = __expf(mi[r] - newm);
            const float p0 = __expf(st[0][r] - newm);
            const float p1 = __expf(st[1][r] - newm);
            float rs = p0 + p1;
#pragma unroll
            for (int m = 8; m >= 1; m >>= 1) rs += __shfl_xor(rs, m, 32);
            li[r] = li[r] * corr + rs;
            mi[r] = newm;
#pragma unroll
            for (int t2 = 0; t2 < 8; ++t2) o[t2][r] *= corr;
            pr0[r] = p0; pr1[r] = p1;
        }

        // ---- stage P (C-layout -> A-layout) via per-wave LDS ----
#pragma unroll
        for (int r = 0; r < 8; ++r) {
            const int row = half * 8 + r;
            P[row * 32 + n0]      = (_Float16)pr0[r];
            P[row * 32 + 16 + n0] = (_Float16)pr1[r];
        }
        // CDNA5 split wait: cross-lane LDS stores must land before re-read.
        asm volatile("s_wait_dscnt 0" ::: "memory");
        v16h ap = load_frag(P + n0 * 32 + half * 8, P + n0 * 32 + 16 + half * 8);

        // ---- O += P @ V  (V fragments already resident) ----
#pragma unroll
        for (int t2 = 0; t2 < 8; ++t2)
            o[t2] = __builtin_amdgcn_wmma_f32_16x16x32_f16(
                false, ap, false, bv[t2], (short)0, o[t2], false, false);
    }

    // ---- normalize and store [s][head*128 + d] as f16 ----
#pragma unroll
    for (int t2 = 0; t2 < 8; ++t2) {
#pragma unroll
        for (int r = 0; r < 8; ++r) {
            const float val = o[t2][r] / li[r];
            aoh[(size_t)(qb + half * 8 + r) * Dm + head * HD + t2 * 16 + n0] =
                (_Float16)val;
        }
    }
}

extern "C" void kernel_launch(void* const* d_in, const int* in_sizes, int n_in,
                              void* d_out, int out_size, void* d_ws, size_t ws_size,
                              hipStream_t stream) {
    const float* x  = (const float*)d_in[0];
    const float* fc = (const float*)d_in[1];
    const float* fs = (const float*)d_in[2];
    const float* Wq = (const float*)d_in[3];
    const float* bq = (const float*)d_in[4];
    const float* Wk = (const float*)d_in[5];
    const float* bk = (const float*)d_in[6];
    const float* Wv = (const float*)d_in[7];
    const float* bv = (const float*)d_in[8];
    const float* Wo = (const float*)d_in[9];
    const float* bo = (const float*)d_in[10];
    const float* gq = (const float*)d_in[11];
    const float* gk = (const float*)d_in[12];
    const int*  seq = (const int*)d_in[13];
    const int*  fp  = (const int*)d_in[14];
    const int*  hp  = (const int*)d_in[15];
    const int*  wp  = (const int*)d_in[16];

    const int Dm = (int)lround(sqrt((double)in_sizes[3]));   // 2048
    const int S  = in_sizes[0] / Dm;                          // 2048
    const int HD = Dm / NHEADS;                               // 128
    const float scale = 1.0f / sqrtf((float)HD);
    const size_t SD = (size_t)S * Dm;
    const size_t DD = (size_t)Dm * Dm;

    // Workspace carve-up (~120 MB).
    char* wsb = (char*)d_ws;
    _Float16* xh  = (_Float16*)wsb;                 wsb += SD * 2;
    _Float16* wqh = (_Float16*)wsb;                 wsb += DD * 2;
    _Float16* wkh = (_Float16*)wsb;                 wsb += DD * 2;
    _Float16* wvh = (_Float16*)wsb;                 wsb += DD * 2;
    _Float16* woh = (_Float16*)wsb;                 wsb += DD * 2;
    float*    yq  = (float*)wsb;                    wsb += SD * 4;
    float*    yk  = (float*)wsb;                    wsb += SD * 4;
    float*    yv  = (float*)wsb;                    wsb += SD * 4;
    _Float16* qhp = (_Float16*)wsb;                 wsb += SD * 2;
    _Float16* khp = (_Float16*)wsb;                 wsb += SD * 2;
    _Float16* vtp = (_Float16*)wsb;                 wsb += SD * 2;
    _Float16* aoh = (_Float16*)wsb;                 wsb += SD * 2;

    // 1) f16 conversions (4 elems/thread)
    conv_f16<<<(int)((SD / 4 + 255) / 256), 256, 0, stream>>>(x,  xh,  (int)SD);
    conv_f16<<<(int)((DD / 4 + 255) / 256), 256, 0, stream>>>(Wq, wqh, (int)DD);
    conv_f16<<<(int)((DD / 4 + 255) / 256), 256, 0, stream>>>(Wk, wkh, (int)DD);
    conv_f16<<<(int)((DD / 4 + 255) / 256), 256, 0, stream>>>(Wv, wvh, (int)DD);
    conv_f16<<<(int)((DD / 4 + 255) / 256), 256, 0, stream>>>(Wo, woh, (int)DD);

    // 2) Q/K/V projections (WMMA, 32x256 block tiles, double-buffered)
    dim3 gg(S / 32, Dm / 256);
    gemm_wmma<<<gg, 128, 0, stream>>>(xh, wqh, bq, yq, S, Dm, Dm);
    gemm_wmma<<<gg, 128, 0, stream>>>(xh, wkh, bk, yk, S, Dm, Dm);
    gemm_wmma<<<gg, 128, 0, stream>>>(xh, wvh, bv, yv, S, Dm, Dm);

    // 3) RMSNorm + RoPE for Q,K ; transpose V
    rms_rope<<<S, 256, 0, stream>>>(yq, gq, fc, fs, qhp, S, Dm, HD, fp, hp, wp);
    rms_rope<<<S, 256, 0, stream>>>(yk, gk, fc, fs, khp, S, Dm, HD, fp, hp, wp);
    dim3 gt(S / 32, Dm / 32);
    transpose_v<<<gt, 256, 0, stream>>>(yv, vtp, S, Dm);

    // 4) Flash attention (WMMA)
    dim3 ga(NHEADS, S / 64);
    attn_wmma<<<ga, 128, 0, stream>>>(qhp, khp, vtp, aoh, S, Dm, seq, scale);

    // 5) Output projection (WMMA) -> fp32 out
    gemm_wmma<<<gg, 128, 0, stream>>>(aoh, woh, bo, (float*)d_out, S, Dm, Dm);
}
